// MultiHeadAttention_11982958755997
// MI455X (gfx1250) — matrix operand order repaired and checked by reading, not compiled, not run on hardware
//
#include <hip/hip_runtime.h>
#include <hip/hip_bf16.h>

typedef __attribute__((ext_vector_type(16))) __bf16 v16bf;
typedef __attribute__((ext_vector_type(8)))  float  v8f;

#define LDT 72  // attention tiles: 64-wide panels (+pad), rows 16B aligned
#define LDK 40  // gemm tiles: 32-wide K panels (+pad), rows 16B aligned

__device__ __forceinline__ v8f wmma_bf16(v16bf a, v16bf b, v8f c) {
  return __builtin_amdgcn_wmma_f32_16x16x32_bf16(false, a, false, b, (short)0, c,
                                                 false, false);
}

// A-fragment (16x32 bf16, MxK). Tile row-major [row][k].
__device__ __forceinline__ v16bf frag_load_a(const __bf16* base, int ld) {
  const int lane = threadIdx.x & 31;
  const __bf16* p = base + (lane & 15) * ld + ((lane >> 4) << 4);
  v16bf f;
#pragma unroll
  for (int e = 0; e < 8; ++e) f[e] = p[e];
#pragma unroll
  for (int e = 0; e < 8; ++e) f[8 + e] = p[8 + e];
  return f;
}

// B-fragment (32x16 bf16, KxN). Tile stored as B^T row-major [n][k].
__device__ __forceinline__ v16bf frag_load_b(const __bf16* base, int ld) {
  const int lane = threadIdx.x & 31;
  const __bf16* p = base + (lane & 15) * ld + ((lane >> 4) << 4);
  v16bf f;
#pragma unroll
  for (int e = 0; e < 16; ++e) f[e] = p[e];
  return f;
}

// CDNA5 async copy: 16B/lane global -> LDS, tracked by ASYNCcnt (no VGPR data).
__device__ __forceinline__ void async_g2l_b128(unsigned lds_off,
                                               const void* gaddr) {
  asm volatile("global_load_async_to_lds_b128 %0, %1, off" ::"v"(lds_off),
               "v"((unsigned long long)(size_t)gaddr)
               : "memory");
}
__device__ __forceinline__ void wait_async0() {
  asm volatile("s_wait_asynccnt 0" ::: "memory");
}

// ---------------------------------------------------------------------------
// C[M,N] = alpha * (A[M,K] @ W[N,K]^T) + bias
// 256 thr (8 waves), block tile 128x128, wave tile 64x32 (4x2 wmma frags).
// ---------------------------------------------------------------------------
template <bool A_BF16, bool OUT_BF16>
__global__ __launch_bounds__(256) void gemm_nt_wmma(
    const void* __restrict__ Av, const float* __restrict__ W,
    const float* __restrict__ bias, void* __restrict__ Outv, int M, int N,
    int K, float alpha) {
  __shared__ __bf16 sA[128 * LDK];
  __shared__ __bf16 sB[128 * LDK];

  const int tid = threadIdx.x;
  const int lane = tid & 31;
  const int wave = tid >> 5;
  const int tileM = blockIdx.y * 128;
  const int tileN = blockIdx.x * 128;
  const int wm = (wave & 1) * 64;   // 0 / 64
  const int wn = (wave >> 1) * 32;  // 0 / 32 / 64 / 96

  v8f acc[4][2];
#pragma unroll
  for (int i = 0; i < 4; ++i)
#pragma unroll
    for (int j = 0; j < 2; ++j)
      acc[i][j] = (v8f){0.f, 0.f, 0.f, 0.f, 0.f, 0.f, 0.f, 0.f};

  const int rr = tid >> 3;       // 0..31
  const int cc = (tid & 7) * 4;  // 0..28

  for (int k0 = 0; k0 < K; k0 += 32) {
#pragma unroll
    for (int i = 0; i < 4; ++i) {
      const int row = rr + i * 32;
      if (A_BF16) {
        const __bf16* g = (const __bf16*)Av + (size_t)(tileM + row) * K + k0 + cc;
#pragma unroll
        for (int j = 0; j < 4; ++j) sA[row * LDK + cc + j] = g[j];
        if (k0 + 32 < K) __builtin_prefetch(g + 32, 0, 1);
      } else {
        const float* g = (const float*)Av + (size_t)(tileM + row) * K + k0 + cc;
#pragma unroll
        for (int j = 0; j < 4; ++j) sA[row * LDK + cc + j] = (__bf16)g[j];
        if (k0 + 32 < K) __builtin_prefetch(g + 32, 0, 1);
      }
      const float* gw = W + (size_t)(tileN + row) * K + k0 + cc;
#pragma unroll
      for (int j = 0; j < 4; ++j) sB[row * LDK + cc + j] = (__bf16)gw[j];
      if (k0 + 32 < K) __builtin_prefetch(gw + 32, 0, 1);
    }
    __syncthreads();

    v16bf a[4], b[2];
#pragma unroll
    for (int mi = 0; mi < 4; ++mi)
      a[mi] = frag_load_a(&sA[(wm + mi * 16) * LDK], LDK);
#pragma unroll
    for (int ni = 0; ni < 2; ++ni)
      b[ni] = frag_load_b(&sB[(wn + ni * 16) * LDK], LDK);
#pragma unroll
    for (int mi = 0; mi < 4; ++mi)
#pragma unroll
      for (int ni = 0; ni < 2; ++ni)
        acc[mi][ni] = wmma_bf16(a[mi], b[ni], acc[mi][ni]);
    __syncthreads();
  }

  const int hi = lane >> 4;
  const int nl = lane & 15;
#pragma unroll
  for (int mi = 0; mi < 4; ++mi)
#pragma unroll
    for (int ni = 0; ni < 2; ++ni)
#pragma unroll
      for (int r = 0; r < 8; ++r) {
        const int gm = tileM + wm + mi * 16 + r + hi * 8;
        const int gn = tileN + wn + ni * 16 + nl;
        float v = acc[mi][ni][r] * alpha;
        if (bias) v += bias[gn];
        if (OUT_BF16)
          ((__bf16*)Outv)[(size_t)gm * N + gn] = (__bf16)v;
        else
          ((float*)Outv)[(size_t)gm * N + gn] = v;
      }
}

// ---------------------------------------------------------------------------
// Flash attention over one (b,h) and a 64-row q-tile. d_k = 64.
// Q/K tiles staged with CDNA5 async global->LDS copies (ASYNCcnt).
// ---------------------------------------------------------------------------
__global__ __launch_bounds__(128) void attn_kernel(
    const __bf16* __restrict__ Qp, const __bf16* __restrict__ Kp,
    const __bf16* __restrict__ Vp, const int* __restrict__ mask,
    __bf16* __restrict__ Out, int Bsz, int S, int D, int H) {
  const int DK = D / H;  // 64
  __shared__ __bf16 sQ[64 * LDT];
  __shared__ __bf16 sK[64 * LDT];   // [key j][d]
  __shared__ __bf16 sVt[64 * LDT];  // transposed: [d][key j]
  __shared__ __bf16 sP[64 * LDT];   // [q row][key j]

  const int tid = threadIdx.x;
  const int lane = tid & 31;
  const int wave = tid >> 5;
  const int hi = lane >> 4;
  const int nl = lane & 15;
  const int bh = blockIdx.y;
  const int b = bh / H, h = bh % H;
  const int q0 = blockIdx.x * 64;

  const __bf16* Qb = Qp + (size_t)b * S * D + h * DK;
  const __bf16* Kb = Kp + (size_t)b * S * D + h * DK;
  const __bf16* Vb = Vp + (size_t)b * S * D + h * DK;

  const int rr = tid >> 3;       // 0..15
  const int cc = (tid & 7) * 8;  // 0..56 (16B chunks)

  // async-stage Q tile once (64x64 bf16)
#pragma unroll
  for (int i = 0; i < 4; ++i) {
    const int row = rr + i * 16;
    async_g2l_b128((unsigned)(size_t)&sQ[row * LDT + cc],
                   Qb + (size_t)(q0 + row) * D + cc);
  }
  wait_async0();
  __syncthreads();

  v16bf aq0 = frag_load_a(&sQ[(wave * 16) * LDT + 0], LDT);
  v16bf aq1 = frag_load_a(&sQ[(wave * 16) * LDT + 32], LDT);

  v8f o[4];
  float rm[8], rl[8];
#pragma unroll
  for (int f = 0; f < 4; ++f)
    o[f] = (v8f){0.f, 0.f, 0.f, 0.f, 0.f, 0.f, 0.f, 0.f};
#pragma unroll
  for (int r = 0; r < 8; ++r) { rm[r] = -3.0e38f; rl[r] = 0.f; }

  for (int kt = 0; kt < S / 64; ++kt) {
    // K tile: async copy. V tile: load + transpose (needs VALU shuffle anyway).
#pragma unroll
    for (int i = 0; i < 4; ++i) {
      const int row = rr + i * 16;
      async_g2l_b128((unsigned)(size_t)&sK[row * LDT + cc],
                     Kb + (size_t)(kt * 64 + row) * D + cc);
      uint4 raw = *(const uint4*)(Vb + (size_t)(kt * 64 + row) * D + cc);
      const __bf16* t = (const __bf16*)&raw;
#pragma unroll
      for (int j = 0; j < 8; ++j) sVt[(cc + j) * LDT + row] = t[j];
      if (kt + 1 < S / 64) {
        __builtin_prefetch(Kb + (size_t)(kt * 64 + 64 + row) * D + cc, 0, 1);
        __builtin_prefetch(Vb + (size_t)(kt * 64 + 64 + row) * D + cc, 0, 1);
      }
    }
    wait_async0();
    __syncthreads();

    // scores S = Q K^T for 4 key sub-tiles of 16
    float sarr[4][8];
#pragma unroll
    for (int jn = 0; jn < 4; ++jn) {
      v16bf bk0 = frag_load_b(&sK[(jn * 16) * LDT + 0], LDT);
      v16bf bk1 = frag_load_b(&sK[(jn * 16) * LDT + 32], LDT);
      v8f s = (v8f){0.f, 0.f, 0.f, 0.f, 0.f, 0.f, 0.f, 0.f};
      s = wmma_bf16(aq0, bk0, s);
      s = wmma_bf16(aq1, bk1, s);
      const int kc = kt * 64 + jn * 16 + nl;
#pragma unroll
      for (int r = 0; r < 8; ++r) {
        const int qr = q0 + wave * 16 + r + hi * 8;
        float sv = s[r];
        if (mask[((size_t)b * S + qr) * S + kc] == 0) sv = -3.0e38f;
        sarr[jn][r] = sv;
      }
    }

    // online softmax
    float mcur[8];
#pragma unroll
    for (int r = 0; r < 8; ++r)
      mcur[r] = fmaxf(fmaxf(sarr[0][r], sarr[1][r]),
                      fmaxf(sarr[2][r], sarr[3][r]));
#pragma unroll
    for (int m = 1; m <= 8; m <<= 1)
#pragma unroll
      for (int r = 0; r < 8; ++r)
        mcur[r] = fmaxf(mcur[r], __shfl_xor(mcur[r], m, 32));

    float esc[8], psum[8];
#pragma unroll
    for (int r = 0; r < 8; ++r) {
      const float mnew = fmaxf(rm[r], mcur[r]);
      esc[r] = __expf(rm[r] - mnew);
      rm[r] = mnew;
      psum[r] = 0.f;
    }
#pragma unroll
    for (int jn = 0; jn < 4; ++jn)
#pragma unroll
      for (int r = 0; r < 8; ++r) {
        const float p = __expf(sarr[jn][r] - rm[r]);
        sarr[jn][r] = p;
        psum[r] += p;
      }
#pragma unroll
    for (int m = 1; m <= 8; m <<= 1)
#pragma unroll
      for (int r = 0; r < 8; ++r) psum[r] += __shfl_xor(psum[r], m, 32);
#pragma unroll
    for (int r = 0; r < 8; ++r) rl[r] = rl[r] * esc[r] + psum[r];
#pragma unroll
    for (int f = 0; f < 4; ++f)
#pragma unroll
      for (int r = 0; r < 8; ++r) o[f][r] *= esc[r];

    // C-layout -> A-layout via LDS (wave-private region, no barrier)
#pragma unroll
    for (int jn = 0; jn < 4; ++jn)
#pragma unroll
      for (int r = 0; r < 8; ++r)
        sP[(wave * 16 + r + hi * 8) * LDT + jn * 16 + nl] = (__bf16)sarr[jn][r];

    v16bf ap0 = frag_load_a(&sP[(wave * 16) * LDT + 0], LDT);
    v16bf ap1 = frag_load_a(&sP[(wave * 16) * LDT + 32], LDT);
#pragma unroll
    for (int dn = 0; dn < 4; ++dn) {
      v16bf bv0 = frag_load_b(&sVt[(dn * 16) * LDT + 0], LDT);
      v16bf bv1 = frag_load_b(&sVt[(dn * 16) * LDT + 32], LDT);
      o[dn] = wmma_bf16(ap0, bv0, o[dn]);
      o[dn] = wmma_bf16(ap1, bv1, o[dn]);
    }
    __syncthreads();
  }

  // normalize + write merged-head output (bf16)
#pragma unroll
  for (int dn = 0; dn < 4; ++dn)
#pragma unroll
    for (int r = 0; r < 8; ++r) {
      const int qr = q0 + wave * 16 + r + hi * 8;
      const int dc = h * DK + dn * 16 + nl;
      Out[((size_t)b * S + qr) * D + dc] = (__bf16)(o[dn][r] / rl[r]);
    }
}

extern "C" void kernel_launch(void* const* d_in, const int* in_sizes, int n_in,
                              void* d_out, int out_size, void* d_ws,
                              size_t ws_size, hipStream_t stream) {
  (void)in_sizes; (void)n_in; (void)out_size; (void)ws_size;
  const float* q   = (const float*)d_in[0];
  const float* k   = (const float*)d_in[1];
  const float* v   = (const float*)d_in[2];
  const int*   msk = (const int*)d_in[3];
  const float* w_q = (const float*)d_in[4];
  const float* w_k = (const float*)d_in[5];
  const float* w_v = (const float*)d_in[6];
  const float* w_o = (const float*)d_in[7];
  const float* b_o = (const float*)d_in[8];

  const int B = 2, S = 2048, D = 1024, H = 16;
  const int M = B * S;               // 4096
  const size_t TEN = (size_t)M * D;  // elems per bf16 tensor

  __bf16* Qp = (__bf16*)d_ws;
  __bf16* Kp = Qp + TEN;
  __bf16* Vp = Kp + TEN;
  __bf16* Ao = Vp + TEN;

  dim3 gg(D / 128, M / 128);  // (8, 32)
  dim3 blk(256);

  gemm_nt_wmma<false, true><<<gg, blk, 0, stream>>>(q, w_q, nullptr, Qp, M, D, D, 0.125f);
  gemm_nt_wmma<false, true><<<gg, blk, 0, stream>>>(k, w_k, nullptr, Kp, M, D, D, 1.0f);
  gemm_nt_wmma<false, true><<<gg, blk, 0, stream>>>(v, w_v, nullptr, Vp, M, D, D, 1.0f);

  attn_kernel<<<dim3(S / 64, B * H), dim3(128), 0, stream>>>(Qp, Kp, Vp, msk, Ao,
                                                             B, S, D, H);

  gemm_nt_wmma<true, false><<<gg, blk, 0, stream>>>(Ao, w_o, b_o, (float*)d_out,
                                                    M, D, D, 1.0f);
}
